// LocalAttention_17927193493963
// MI455X (gfx1250) — matrix operand order repaired and checked by reading, not compiled, not run on hardware
//
#include <hip/hip_runtime.h>

#define BB 4
#define SS 2048
#define DD 1024
#define MM (BB * SS)

typedef __attribute__((ext_vector_type(16))) __bf16 v16bf;
typedef __attribute__((ext_vector_type(8)))  __bf16 v8bf;
typedef __attribute__((ext_vector_type(8)))  float  v8f;
typedef __attribute__((ext_vector_type(4)))  unsigned int v4ui;
typedef __attribute__((ext_vector_type(8)))  int  v8i;
typedef __attribute__((ext_vector_type(4)))  int  v4i;

#if __has_builtin(__builtin_amdgcn_tensor_load_to_lds)
#define USE_TDM 1
#else
#define USE_TDM 0
#endif

static __device__ __forceinline__ float bf2f(__bf16 v) {
  unsigned short h; __builtin_memcpy(&h, &v, 2);
  unsigned u = ((unsigned)h) << 16;
  float f; __builtin_memcpy(&f, &u, 4);
  return f;
}

static __device__ __forceinline__ __bf16 f2bf(float f) {
  unsigned u; __builtin_memcpy(&u, &f, 4);
  unsigned r = u + 0x7FFFu + ((u >> 16) & 1u);   // round-to-nearest-even
  unsigned short h = (unsigned short)(r >> 16);
  __bf16 o; __builtin_memcpy(&o, &h, 2);
  return o;
}

#if USE_TDM
// Issue a TDM load of a [rows x tileK] bf16 tile (row stride = strideElems)
// from global memory into LDS at byte offset ldsOff.  D# per ISA ch.8.
static __device__ __forceinline__ void tdm_load_tile(const __bf16* gsrc, unsigned ldsOff,
                                                     int rows, int tileK, int strideElems) {
  unsigned long long ga = (unsigned long long)(size_t)gsrc;
  v4ui g0 = {};
  g0[0] = 1u;                                            // count=1 (user descriptor)
  g0[1] = ldsOff;                                        // lds_addr (bytes)
  g0[2] = (unsigned)(ga & 0xFFFFFFFFu);                  // global_addr[31:0]
  g0[3] = (unsigned)((ga >> 32) & 0x1FFFFFFu) | (2u << 30);  // global_addr[56:32] | type=2
  v8i g1 = {};
  g1[0] = (int)(1u << 16);                               // workgroup_mask=0, data_size=2B
  g1[1] = (int)(((unsigned)strideElems & 0xFFFFu) << 16);            // tensor_dim0 lo16
  g1[2] = (int)((((unsigned)strideElems >> 16) & 0xFFFFu) |
                (((unsigned)rows & 0xFFFFu) << 16));                 // tensor_dim0 hi | tensor_dim1 lo
  g1[3] = (int)((((unsigned)rows >> 16) & 0xFFFFu) |
                (((unsigned)tileK & 0xFFFFu) << 16));                // tensor_dim1 hi | tile_dim0
  g1[4] = (int)((unsigned)rows & 0xFFFFu);                           // tile_dim1 (tile_dim2=0)
  g1[5] = strideElems;                                               // tensor_dim0_stride lo32
  g1[6] = 0;
  g1[7] = 0;
  v4i z4 = {};
#if __clang_major__ >= 23
  v8i z8 = {};
  __builtin_amdgcn_tensor_load_to_lds(g0, g1, z4, z4, z8, 0);
#else
  __builtin_amdgcn_tensor_load_to_lds(g0, g1, z4, z4, 0);
#endif
}
#endif

// ---------------- elementwise conversion / fill ----------------

__global__ void cvt_f32_bf16(const float* __restrict__ in, __bf16* __restrict__ out, int n) {
  int i = blockIdx.x * blockDim.x + threadIdx.x;
  int stride = gridDim.x * blockDim.x;
  for (; i < n; i += stride) out[i] = f2bf(in[i]);
}

__global__ void fill_zero_f32(float* __restrict__ p, int n4) {
  int i = blockIdx.x * blockDim.x + threadIdx.x;
  int stride = gridDim.x * blockDim.x;
  float4 z = make_float4(0.f, 0.f, 0.f, 0.f);
  for (; i < n4; i += stride) ((float4*)p)[i] = z;
}

// ---------------- NT GEMM: TDM-staged B in LDS + v_wmma_f32_16x16x32_bf16 ----
// C[M,N] = A[M,K] * B[N,K]^T + bias[N].  A,B bf16 row-major; out f32 or bf16.
// Block = 128 threads = 4 waves; each wave owns a 32x32 tile (2x2 WMMA tiles).
// The 64 weight rows used by the block are DMA'd into LDS in 256-wide K-chunks
// (double-buffered) by the Tensor Data Mover; A streams from global.
#define KC 256
template <bool OUT_BF16>
__global__ void __launch_bounds__(128)
gemm_nt_wmma(const __bf16* __restrict__ A, const __bf16* __restrict__ Bw,
             const float* __restrict__ bias, void* __restrict__ Cout,
             int M, int N, int K) {
  __shared__ __bf16 Btile[2][64][KC];          // 2 x 32KB

  const int lane = threadIdx.x & 31;
  const int wave = threadIdx.x >> 5;
  const int l15  = lane & 15;
  const int kh   = lane >> 4;                  // 0 or 1 (lane half)

  const int m0     = blockIdx.y * 64 + (wave >> 1) * 32;
  const int nBlock = blockIdx.x * 64;
  const int n0     = nBlock + (wave & 1) * 32;

  v8f acc[2][2] = {};

  // A fragment: lane (m = l15 [+16]) holds K chunks [kh*8, +8) and [16+kh*8, +8)
  const __bf16* Arow0 = A + (size_t)(m0 + l15) * K;
  const __bf16* Arow1 = Arow0 + (size_t)16 * K;

  const __bf16* Bblk = Bw + (size_t)nBlock * K;
  const int nChunks = K / KC;

#if USE_TDM
  const unsigned ldsBase[2] = { (unsigned)(size_t)&Btile[0][0][0],
                                (unsigned)(size_t)&Btile[1][0][0] };
  if (threadIdx.x == 0)
    tdm_load_tile(Bblk, ldsBase[0], 64, KC, K);
#endif

  for (int c = 0; c < nChunks; ++c) {
    const int buf = c & 1;
#if USE_TDM
    __builtin_amdgcn_s_wait_tensorcnt(0);      // chunk c landed in LDS
    __syncthreads();                           // ...and everyone knows it
    if (threadIdx.x == 0 && c + 1 < nChunks)   // kick chunk c+1 into the other buffer
      tdm_load_tile(Bblk + (c + 1) * KC, ldsBase[(c + 1) & 1], 64, KC, K);
#else
    {
      const __bf16* src = Bblk + c * KC;
      for (int idx = threadIdx.x; idx < 64 * (KC / 8); idx += 128) {
        int r = idx >> 5, cc = idx & 31;
        *(v8bf*)&Btile[buf][r][cc * 8] = *(const v8bf*)(src + (size_t)r * K + cc * 8);
      }
    }
    __syncthreads();
#endif

    // B fragment source rows for this wave, from LDS
    const __bf16* Bl0 = &Btile[buf][(wave & 1) * 32 + l15][0];
    const __bf16* Bl1 = Bl0 + 16 * KC;
    const int kbase = c * KC;

    for (int kk = 0; kk < KC; kk += 32) {
      const int k0 = kbase + kk;
      v8bf a0lo = *(const v8bf*)(Arow0 + k0 + kh * 8);
      v8bf a0hi = *(const v8bf*)(Arow0 + k0 + 16 + kh * 8);
      v8bf a1lo = *(const v8bf*)(Arow1 + k0 + kh * 8);
      v8bf a1hi = *(const v8bf*)(Arow1 + k0 + 16 + kh * 8);
      v16bf afrag0 = __builtin_shufflevector(a0lo, a0hi, 0,1,2,3,4,5,6,7,8,9,10,11,12,13,14,15);
      v16bf afrag1 = __builtin_shufflevector(a1lo, a1hi, 0,1,2,3,4,5,6,7,8,9,10,11,12,13,14,15);

      v16bf bfrag0 = __builtin_shufflevector(*(const v8bf*)(Bl0 + kk + kh * 8), *(const v8bf*)(Bl0 + kk + 16 + kh * 8), 0,1,2,3,4,5,6,7,8,9,10,11,12,13,14,15);
      v16bf bfrag1 = __builtin_shufflevector(*(const v8bf*)(Bl1 + kk + kh * 8), *(const v8bf*)(Bl1 + kk + 16 + kh * 8), 0,1,2,3,4,5,6,7,8,9,10,11,12,13,14,15);

      if (k0 + 32 < K)                          // pull next A tile (global_prefetch)
        __builtin_prefetch(Arow0 + k0 + 32, 0, 1);

      acc[0][0] = __builtin_amdgcn_wmma_f32_16x16x32_bf16(false, afrag0, false, bfrag0,
                                                          (short)0, acc[0][0], false, false);
      acc[0][1] = __builtin_amdgcn_wmma_f32_16x16x32_bf16(false, afrag0, false, bfrag1,
                                                          (short)0, acc[0][1], false, false);
      acc[1][0] = __builtin_amdgcn_wmma_f32_16x16x32_bf16(false, afrag1, false, bfrag0,
                                                          (short)0, acc[1][0], false, false);
      acc[1][1] = __builtin_amdgcn_wmma_f32_16x16x32_bf16(false, afrag1, false, bfrag1,
                                                          (short)0, acc[1][1], false, false);
    }
    __syncthreads();                            // buffer free for reuse
  }

  // C/D layout: lane L -> n = n0+(L&15), rows m0+(L>>4)*8+r (r = VGPR index)
#pragma unroll
  for (int ii = 0; ii < 2; ++ii) {
#pragma unroll
    for (int jj = 0; jj < 2; ++jj) {
      int n  = n0 + jj * 16 + l15;
      int mb = m0 + ii * 16 + kh * 8;
      float bv = bias[n];
#pragma unroll
      for (int r = 0; r < 8; ++r) {
        float v = acc[ii][jj][r] + bv;
        if (OUT_BF16) ((__bf16*)Cout)[(size_t)(mb + r) * N + n] = f2bf(v);
        else          ((float*)Cout)[(size_t)(mb + r) * N + n] = v;
      }
    }
  }
}

// ---------------- banded attention: one wave per query ----------------
__global__ void __launch_bounds__(256)
local_attn(const __bf16* __restrict__ Qb, const __bf16* __restrict__ Kb,
           const __bf16* __restrict__ Vb, float* __restrict__ attn,
           __bf16* __restrict__ ctx, float scale) {
  const int lane = threadIdx.x & 31;
  const int wave = threadIdx.x >> 5;
  const int q = blockIdx.x * 8 + wave;       // global query over B*S
  if (q >= MM) return;
  const int b  = q / SS;
  const int qi = q - b * SS;

  const __bf16* Qrow = Qb + (size_t)q * DD;
  const __bf16* Kbat = Kb + (size_t)b * SS * DD;
  const __bf16* Vbat = Vb + (size_t)b * SS * DD;

  float s[11];
#pragma unroll
  for (int jj = 0; jj < 11; ++jj) {
    int j = qi - 5 + jj;
    float acc = -1e30f;
    if (j >= 0 && j < SS) {
      const __bf16* Krow = Kbat + (size_t)j * DD;
      acc = 0.f;
      for (int t = 0; t < DD / 256; ++t) {      // 8 bf16 per lane per step
        int d0 = (t * 32 + lane) * 8;
        v8bf qa = *(const v8bf*)(Qrow + d0);
        v8bf ka = *(const v8bf*)(Krow + d0);
#pragma unroll
        for (int e = 0; e < 8; ++e) acc += bf2f(qa[e]) * bf2f(ka[e]);
      }
      for (int off = 16; off > 0; off >>= 1) acc += __shfl_down(acc, off, 32);
      acc = __shfl(acc, 0, 32) * scale;
    }
    s[jj] = acc;
  }

  float mx = s[0];
#pragma unroll
  for (int jj = 1; jj < 11; ++jj) mx = fmaxf(mx, s[jj]);
  float sum = 0.f;
#pragma unroll
  for (int jj = 0; jj < 11; ++jj) { s[jj] = __expf(s[jj] - mx); sum += s[jj]; }
  float inv = 1.f / sum;
#pragma unroll
  for (int jj = 0; jj < 11; ++jj) s[jj] *= inv;

  if (lane == 0) {
#pragma unroll
    for (int jj = 0; jj < 11; ++jj) {
      int j = qi - 5 + jj;
      if (j >= 0 && j < SS) attn[(size_t)q * SS + j] = s[jj];
    }
  }

  for (int t = 0; t < DD / 32; ++t) {
    int d = t * 32 + lane;
    float c = 0.f;
#pragma unroll
    for (int jj = 0; jj < 11; ++jj) {
      int j = qi - 5 + jj;
      if (j >= 0 && j < SS) c += s[jj] * bf2f(Vbat[(size_t)j * DD + d]);
    }
    ctx[(size_t)q * DD + d] = f2bf(c);
  }
}

// ---------------- launcher ----------------
extern "C" void kernel_launch(void* const* d_in, const int* in_sizes, int n_in,
                              void* d_out, int out_size, void* d_ws, size_t ws_size,
                              hipStream_t stream) {
  (void)in_sizes; (void)n_in; (void)out_size; (void)ws_size;

  const float* x  = (const float*)d_in[0];
  const float* Wq = (const float*)d_in[1];
  const float* bq = (const float*)d_in[2];
  const float* Wk = (const float*)d_in[3];
  const float* bk = (const float*)d_in[4];
  const float* Wv = (const float*)d_in[5];
  const float* bv = (const float*)d_in[6];
  const float* Wo = (const float*)d_in[7];
  const float* bo = (const float*)d_in[8];

  float* out  = (float*)d_out;                  // [B,S,D]
  float* attn = out + (size_t)MM * DD;          // [B,S,S]

  char* ws = (char*)d_ws;
  size_t off = 0;
  auto take = [&](size_t bytes) -> char* {
    char* p = ws + off;
    off += (bytes + 255) & ~(size_t)255;
    return p;
  };
  __bf16* xb   = (__bf16*)take((size_t)MM * DD * 2);
  __bf16* wqb  = (__bf16*)take((size_t)DD * DD * 2);
  __bf16* wkb  = (__bf16*)take((size_t)DD * DD * 2);
  __bf16* wvb  = (__bf16*)take((size_t)DD * DD * 2);
  __bf16* wob  = (__bf16*)take((size_t)DD * DD * 2);
  __bf16* Qb   = (__bf16*)take((size_t)MM * DD * 2);
  __bf16* Kbuf = (__bf16*)take((size_t)MM * DD * 2);
  __bf16* Vbuf = (__bf16*)take((size_t)MM * DD * 2);
  __bf16* ctxb = (__bf16*)take((size_t)MM * DD * 2);

  // 1) fp32 -> bf16
  cvt_f32_bf16<<<1024, 256, 0, stream>>>(x,  xb,  MM * DD);
  cvt_f32_bf16<<<256,  256, 0, stream>>>(Wq, wqb, DD * DD);
  cvt_f32_bf16<<<256,  256, 0, stream>>>(Wk, wkb, DD * DD);
  cvt_f32_bf16<<<256,  256, 0, stream>>>(Wv, wvb, DD * DD);
  cvt_f32_bf16<<<256,  256, 0, stream>>>(Wo, wob, DD * DD);

  // 2) Q/K/V projections (TDM-staged weights, WMMA bf16, f32 accumulate)
  dim3 gg(DD / 64, MM / 64);   // (16, 128)
  gemm_nt_wmma<true><<<gg, 128, 0, stream>>>(xb, wqb, bq, Qb,   MM, DD, DD);
  gemm_nt_wmma<true><<<gg, 128, 0, stream>>>(xb, wkb, bk, Kbuf, MM, DD, DD);
  gemm_nt_wmma<true><<<gg, 128, 0, stream>>>(xb, wvb, bv, Vbuf, MM, DD, DD);

  // 3) attn output is a [B,S,S] banded matrix: zero it, then fill the band
  fill_zero_f32<<<2048, 256, 0, stream>>>(attn, (int)(((size_t)MM * SS) / 4));
  local_attn<<<MM / 8, 256, 0, stream>>>(Qb, Kbuf, Vbuf, attn, ctxb, 0.03125f /*1/sqrt(1024)*/);

  // 4) output projection (WMMA, f32 store into d_out)
  gemm_nt_wmma<false><<<gg, 128, 0, stream>>>(ctxb, wob, bo, out, MM, DD, DD);
}
